// TimeMoEAttention_5677946765399
// MI455X (gfx1250) — compile-verified, matrix operand order checked by reading, not run on hardware
//
#include <hip/hip_runtime.h>
#include <hip/hip_bf16.h>
#include <math.h>

// ---------------- model dims ----------------
#define BB   16
#define TT   512
#define DD   256
#define EE   12
#define HH2  64          // 2*H
#define NHD  4           // heads
#define HDIM 64          // head dim
#define OO   64
#define BT   (BB * TT)   // 8192

typedef __attribute__((ext_vector_type(16))) __bf16 v16bf;
typedef __attribute__((ext_vector_type(8)))  __bf16 v8bf;
typedef __attribute__((ext_vector_type(8)))  float  v8f;

__device__ __forceinline__ v8f wmma_bf16(v16bf a, v16bf b, v8f c) {
    // D = A(16x32 bf16) * B(32x16 bf16) + C(16x16 f32)
    return __builtin_amdgcn_wmma_f32_16x16x32_bf16(
        false, a, false, b, (short)0, c, false, false);
}

// Unified fragment loader (ISA 7.12.2): works for A (row = M) and for B when B
// is stored N-major [N, K] (row = N).  lane&15 selects row; per lane the K
// pattern is 8 contiguous elements at k0+8*(lane>=16) and 8 more at +16.
// With bf16 data this is exactly two 16-byte loads (ds_load_b128 /
// global_load_b128).
__device__ __forceinline__ v16bf load_frag(const __bf16* base, int ld, int k0, int lane) {
    int r  = lane & 15;
    int hi = (lane >> 4) & 1;
    const __bf16* p = base + (size_t)r * ld + k0 + hi * 8;
    v8bf lo = *(const v8bf*)p;
    v8bf hh = *(const v8bf*)(p + 16);
    return __builtin_shufflevector(lo, hh, 0, 1, 2, 3, 4, 5, 6, 7,
                                           8, 9, 10, 11, 12, 13, 14, 15);
}

// ---- CDNA5 async global->LDS copy (16B per lane), tracked on ASYNCcnt ----
__device__ __forceinline__ void lds_async_b128(void* lds_dst, const void* gsrc) {
    unsigned lds_off = (unsigned)(unsigned long long)lds_dst;  // addr[31:0] = LDS offset
    asm volatile("global_load_async_to_lds_b128 %0, %1, off"
                 :: "v"(lds_off), "v"(gsrc) : "memory");
}
__device__ __forceinline__ void wait_async0() {
    asm volatile("s_wait_asynccnt 0" ::: "memory");
}

__device__ __forceinline__ float wave_sum(float v) {
#pragma unroll
    for (int o = 16; o; o >>= 1) v += __shfl_xor(v, o, 32);
    return v;
}
__device__ __forceinline__ float wave_max(float v) {
#pragma unroll
    for (int o = 16; o; o >>= 1) v = fmaxf(v, __shfl_xor(v, o, 32));
    return v;
}

// =================== repack helpers ===================
// transpose + convert: src [nmat, R, C] f32  ->  dst [nmat, C, R] bf16
__global__ void tcvt_kernel(const float* __restrict__ src, __bf16* __restrict__ dst,
                            int R, int C) {
    int m = blockIdx.y;
    const float* s = src + (size_t)m * R * C;
    __bf16* d = dst + (size_t)m * R * C;
    int i = blockIdx.x * 256 + threadIdx.x;
    if (i < R * C) {
        int r = i / C, c = i % C;
        d[(size_t)c * R + r] = (__bf16)s[i];
    }
}

// straight f32 -> bf16
__global__ void cvt_kernel(const float* __restrict__ src, __bf16* __restrict__ dst) {
    int i = blockIdx.x * 256 + threadIdx.x;
    dst[i] = (__bf16)src[i];
}

// q,k -> bf16 copies; v -> bf16 transposed per (b,head): vt[b][h][feat][t]
__global__ void pack_attn_kernel(const float* __restrict__ q, const float* __restrict__ k,
                                 const float* __restrict__ v,
                                 __bf16* __restrict__ qb, __bf16* __restrict__ kb,
                                 __bf16* __restrict__ vt) {
    int i = blockIdx.x * 256 + threadIdx.x;   // over BT*D
    qb[i] = (__bf16)q[i];
    kb[i] = (__bf16)k[i];
    int d = i & (DD - 1), row = i >> 8;
    int t = row & (TT - 1), b = row >> 9;
    int h = d >> 6, f = d & 63;
    vt[(((size_t)(b * NHD + h) * HDIM + f) * TT) + t] = (__bf16)v[i];
}

// =================== router: combine weights [BT, E] ===================
__global__ void moe_combine_kernel(const float* __restrict__ x,
                                   const float* __restrict__ gate,   // [D, E]
                                   float* __restrict__ combine) {    // [BT, E]
    int tid = threadIdx.x, wave = tid >> 5, lane = tid & 31;
    int t = blockIdx.x * 8 + wave;
    const float* xr = x + (size_t)t * DD;
    float lg[EE];
#pragma unroll
    for (int e = 0; e < EE; ++e) lg[e] = 0.f;
#pragma unroll
    for (int i = 0; i < DD / 32; ++i) {
        float xv = xr[lane + i * 32];
        const float* g = gate + (size_t)(lane + i * 32) * EE;
#pragma unroll
        for (int e = 0; e < EE; ++e) lg[e] += xv * g[e];
    }
#pragma unroll
    for (int e = 0; e < EE; ++e) lg[e] = wave_sum(lg[e]);
    if (lane == 0) {
        int i1 = 0; float m1 = lg[0];
        for (int e = 1; e < EE; ++e) if (lg[e] > m1) { m1 = lg[e]; i1 = e; }
        int i2 = -1; float m2 = -1e30f;
        for (int e = 0; e < EE; ++e) if (e != i1 && lg[e] > m2) { m2 = lg[e]; i2 = e; }
        float e2 = __expf(m2 - m1);
        float inv = 1.0f / (1.0f + e2);
        float* crow = combine + (size_t)t * EE;
#pragma unroll
        for (int e = 0; e < EE; ++e) crow[e] = 0.f;
        crow[i1] = inv;
        crow[i2] = e2 * inv;
    }
}

// =================== MoE experts (dense) ===================
// grid = BT/16, block 128 (4 waves).  Weights pre-transposed bf16:
//   w1t [E, 2H, D]  (N-major),  w2t [E, D, 2H]  (N-major)
__global__ void moe_expert_kernel(const __bf16* __restrict__ xb,     // [BT, D] bf16
                                  const __bf16* __restrict__ w1t,
                                  const __bf16* __restrict__ w2t,
                                  const float* __restrict__ combine, // [BT, E]
                                  float* __restrict__ y) {           // [BT, D]
    __shared__ __align__(16) __bf16 xt[16 * DD];   // 8 KB
    __shared__ __align__(16) __bf16 ht[16 * HH2];  // 2 KB
    const int tid = threadIdx.x, wave = tid >> 5, lane = tid & 31;
    const int hi = (lane >> 4) & 1, nn = lane & 15;
    const int row0 = blockIdx.x * 16;

    // async-stage the contiguous 8KB bf16 tile into LDS (4 x 16B per thread)
    {
        const __bf16* gsrc = xb + (size_t)row0 * DD;
#pragma unroll
        for (int j = 0; j < 4; ++j) {
            int c = (tid + j * 128) * 8;          // element offset (8 bf16 = 16B)
            lds_async_b128(&xt[c], gsrc + c);
        }
        wait_async0();
    }
    __syncthreads();

    v8f acc[4] = {};   // wave owns D-cols [wave*64, wave*64+64)

    for (int e = 0; e < EE; ++e) {
        // ---- stage 1: h = gelu(x @ w1[e]); wave computes h cols wave*16.. ----
        v8f c1 = {};
        const __bf16* b1 = w1t + ((size_t)e * HH2 + wave * 16) * DD;
#pragma unroll
        for (int k0 = 0; k0 < DD; k0 += 32) {
            v16bf a = load_frag(xt, DD, k0, lane);
            v16bf b = load_frag(b1, DD, k0, lane);
            c1 = wmma_bf16(a, b, c1);
        }
#pragma unroll
        for (int r = 0; r < 8; ++r) {
            float xv = c1[r];
            float g = 0.5f * xv * (1.0f + erff(xv * 0.70710678118f)); // exact GELU
            ht[(r + 8 * hi) * HH2 + wave * 16 + nn] = (__bf16)g;
        }
        __syncthreads();

        // ---- stage 2: acc += combine[:,e] * (h @ w2[e]) ----
        float cwv[8];
#pragma unroll
        for (int r = 0; r < 8; ++r)
            cwv[r] = combine[(size_t)(row0 + r + 8 * hi) * EE + e];
#pragma unroll
        for (int nt = 0; nt < 4; ++nt) {
            v8f d = {};
            const __bf16* b2 = w2t + ((size_t)e * DD + wave * 64 + nt * 16) * HH2;
#pragma unroll
            for (int k0 = 0; k0 < HH2; k0 += 32) {
                v16bf a = load_frag(ht, HH2, k0, lane);
                v16bf b = load_frag(b2, HH2, k0, lane);
                d = wmma_bf16(a, b, d);
            }
#pragma unroll
            for (int r = 0; r < 8; ++r) acc[nt][r] += cwv[r] * d[r];
        }
        __syncthreads();  // ht reused next expert
    }

#pragma unroll
    for (int nt = 0; nt < 4; ++nt)
#pragma unroll
        for (int r = 0; r < 8; ++r)
            y[(size_t)(row0 + r + 8 * hi) * DD + wave * 64 + nt * 16 + nn] = acc[nt][r];
}

// =================== multi-head attention ===================
// grid = B*NH*(T/16) = 2048, block 128 (4 waves)
// qb, kb: [BT, D] bf16 ; vt: [B, NH, hd, T] bf16
__global__ void attention_kernel(const __bf16* __restrict__ qb,
                                 const __bf16* __restrict__ kb,
                                 const __bf16* __restrict__ vt,
                                 float* __restrict__ out) {
    __shared__ __align__(16) float  sc[16 * TT];     // 32 KB score rows (f32)
    __shared__ __align__(16) __bf16 pb[16 * TT];     // 16 KB probs (bf16)
    __shared__ __align__(16) __bf16 qt[16 * HDIM];   // 2 KB
    int blk  = blockIdx.x;
    int qt16 = blk & 31;               // T/16 = 32
    int head = (blk >> 5) & 3;
    int b    = blk >> 7;
    int tid = threadIdx.x, wave = tid >> 5, lane = tid & 31;
    const int hi = (lane >> 4) & 1, nn = lane & 15;
    const size_t base = ((size_t)b * TT) * DD + head * HDIM;
    const int qrow0 = qt16 * 16;

    // async-stage Q tile (16 rows x 64 bf16): 1 x 16B chunk per thread
    {
        int r = tid >> 3, c = (tid & 7) * 8;
        lds_async_b128(&qt[r * HDIM + c],
                       qb + base + (size_t)(qrow0 + r) * DD + c);
        wait_async0();
    }
    __syncthreads();

    // scores = (Q @ K^T) * 1/sqrt(hd)
    for (int kt = wave; kt < TT / 16; kt += 4) {
        v8f c = {};
        const __bf16* kbase = kb + base + (size_t)(kt * 16) * DD;  // row n = key
#pragma unroll
        for (int k0 = 0; k0 < HDIM; k0 += 32) {
            v16bf a  = load_frag(qt, HDIM, k0, lane);
            v16bf bf = load_frag(kbase, DD, k0, lane);
            c = wmma_bf16(a, bf, c);
        }
#pragma unroll
        for (int r = 0; r < 8; ++r)
            sc[(r + 8 * hi) * TT + kt * 16 + nn] = c[r] * 0.125f;
    }
    __syncthreads();

    // fp32 softmax per row; normalized probabilities written as bf16
    for (int rr = 0; rr < 4; ++rr) {
        int row = wave * 4 + rr;
        float* srow = sc + row * TT;
        __bf16* prow = pb + row * TT;
        float mx = -1e30f;
        for (int j = lane; j < TT; j += 32) mx = fmaxf(mx, srow[j]);
        mx = wave_max(mx);
        float sm = 0.f;
        for (int j = lane; j < TT; j += 32) {
            float ev = __expf(srow[j] - mx);
            srow[j] = ev; sm += ev;
        }
        sm = wave_sum(sm);
        float inv = 1.0f / sm;
        for (int j = lane; j < TT; j += 32) prow[j] = (__bf16)(srow[j] * inv);
    }
    __syncthreads();

    // out = P @ V : wave owns feature cols wave*16..
    v8f c = {};
    const __bf16* vbase = vt + ((size_t)(b * NHD + head) * HDIM + wave * 16) * TT;
#pragma unroll
    for (int k0 = 0; k0 < TT; k0 += 32) {
        v16bf a  = load_frag(pb, TT, k0, lane);
        v16bf bf = load_frag(vbase, TT, k0, lane);
        c = wmma_bf16(a, bf, c);
    }
#pragma unroll
    for (int r = 0; r < 8; ++r)
        out[base + (size_t)(qrow0 + r + 8 * hi) * DD + wave * 16 + nn] = c[r];
}

// =================== QKV projection GEMM ===================
// grid = (BT/16, D/64, 3), block 128.  Wt: [3, D(out), D(in)] bf16 N-major.
__global__ void qkv_gemm_kernel(const __bf16* __restrict__ Ab,  // [BT, D] bf16
                                const __bf16* __restrict__ Wt,
                                float* __restrict__ out) {
    __shared__ __align__(16) __bf16 at[16 * DD];  // 8 KB
    int tile = blockIdx.x, nblk = blockIdx.y, j = blockIdx.z;
    int tid = threadIdx.x, wave = tid >> 5, lane = tid & 31;
    const int hi = (lane >> 4) & 1, nn = lane & 15;
    const int row0 = tile * 16;

    {
        const __bf16* gsrc = Ab + (size_t)row0 * DD;
#pragma unroll
        for (int jj = 0; jj < 4; ++jj) {
            int c = (tid + jj * 128) * 8;
            lds_async_b128(&at[c], gsrc + c);
        }
        wait_async0();
    }
    __syncthreads();

    int ncol = nblk * 64 + wave * 16;
    const __bf16* bbase = Wt + ((size_t)j * DD + ncol) * DD;
    float* oj = out + (size_t)j * BT * DD;
    v8f c = {};
#pragma unroll
    for (int k0 = 0; k0 < DD; k0 += 32) {
        v16bf a = load_frag(at, DD, k0, lane);
        v16bf b = load_frag(bbase, DD, k0, lane);
        c = wmma_bf16(a, b, c);
    }
#pragma unroll
    for (int r = 0; r < 8; ++r)
        oj[(size_t)(row0 + r + 8 * hi) * DD + ncol + nn] = c[r];
}

// =================== LayerNorm over D=256 (f32 + bf16 outputs) ===============
__global__ void ln_kernel(const float* __restrict__ x, const float* __restrict__ s,
                          const float* __restrict__ bvec,
                          float* __restrict__ y, __bf16* __restrict__ yb) {
    int tid = threadIdx.x, wave = tid >> 5, lane = tid & 31;
    int row = blockIdx.x * 8 + wave;
    const float* xr = x + (size_t)row * DD;
    float vals[DD / 32];
    float sum = 0.f;
#pragma unroll
    for (int i = 0; i < DD / 32; ++i) { vals[i] = xr[lane + i * 32]; sum += vals[i]; }
    float mean = wave_sum(sum) * (1.0f / DD);
    float vs = 0.f;
#pragma unroll
    for (int i = 0; i < DD / 32; ++i) { float d = vals[i] - mean; vs += d * d; }
    float var = wave_sum(vs) * (1.0f / DD);
    float rs = rsqrtf(var + 1e-5f);
#pragma unroll
    for (int i = 0; i < DD / 32; ++i) {
        int c = lane + i * 32;
        float o = (vals[i] - mean) * rs * s[c] + bvec[c];
        y[(size_t)row * DD + c] = o;
        yb[(size_t)row * DD + c] = (__bf16)o;
    }
}

// =================== residual add ===================
__global__ void add_kernel(const float* __restrict__ a, const float* __restrict__ b,
                           float* __restrict__ o, int n) {
    int i = blockIdx.x * blockDim.x + threadIdx.x;
    if (i < n) o[i] = a[i] + b[i];
}

// =================== final head: mean_T -> GEMM -> LN ===================
// single block of 128;  outwt: [O, D] bf16 N-major
__global__ void final_head_kernel(const float* __restrict__ h,
                                  const __bf16* __restrict__ outwt,
                                  const float* __restrict__ lno_s,
                                  const float* __restrict__ lno_b,
                                  float* __restrict__ outp) {       // [B, O]
    __shared__ __align__(16) __bf16 hm[BB * DD];
    __shared__ __align__(16) float  res[BB * OO];
    int tid = threadIdx.x, wave = tid >> 5, lane = tid & 31;
    const int hi = (lane >> 4) & 1, nn = lane & 15;

    for (int i = tid; i < BB * DD; i += 128) {
        int b = i >> 8, d = i & (DD - 1);
        const float* p = h + ((size_t)b * TT) * DD + d;
        float sacc = 0.f;
        for (int t = 0; t < TT; ++t) sacc += p[(size_t)t * DD];
        hm[i] = (__bf16)(sacc * (1.0f / TT));
    }
    __syncthreads();

    v8f c = {};
    const __bf16* bbase = outwt + (size_t)(wave * 16) * DD;
#pragma unroll
    for (int k0 = 0; k0 < DD; k0 += 32) {
        v16bf a = load_frag(hm, DD, k0, lane);
        v16bf b = load_frag(bbase, DD, k0, lane);
        c = wmma_bf16(a, b, c);
    }
#pragma unroll
    for (int r = 0; r < 8; ++r)
        res[(r + 8 * hi) * OO + wave * 16 + nn] = c[r];
    __syncthreads();

    if (tid < BB) {
        float m = 0.f;
        for (int jx = 0; jx < OO; ++jx) m += res[tid * OO + jx];
        m *= (1.0f / OO);
        float vv = 0.f;
        for (int jx = 0; jx < OO; ++jx) { float d = res[tid * OO + jx] - m; vv += d * d; }
        vv *= (1.0f / OO);
        float rs = rsqrtf(vv + 1e-5f);
        for (int jx = 0; jx < OO; ++jx)
            outp[tid * OO + jx] = (res[tid * OO + jx] - m) * rs * lno_s[jx] + lno_b[jx];
    }
}

// =================== orchestration ===================
extern "C" void kernel_launch(void* const* d_in, const int* in_sizes, int n_in,
                              void* d_out, int out_size, void* d_ws, size_t ws_size,
                              hipStream_t stream) {
    (void)in_sizes; (void)n_in; (void)out_size; (void)ws_size;
    const float* x        = (const float*)d_in[0];
    const float* gate_qkv = (const float*)d_in[1];   // [3, D, E]
    const float* w1_qkv   = (const float*)d_in[2];   // [3, E, D, 2H]
    const float* w2_qkv   = (const float*)d_in[3];   // [3, E, 2H, D]
    const float* ln1_s    = (const float*)d_in[4];
    const float* ln1_b    = (const float*)d_in[5];
    const float* ln2_s    = (const float*)d_in[6];
    const float* ln2_b    = (const float*)d_in[7];
    const float* qkv_w    = (const float*)d_in[8];   // [L, 3, D, D]
    const float* gate_b   = (const float*)d_in[9];   // [L, D, E]
    const float* w1_b     = (const float*)d_in[10];  // [L, E, D, 2H]
    const float* w2_b     = (const float*)d_in[11];  // [L, E, 2H, D]
    const float* out_w    = (const float*)d_in[12];  // [D, O]
    const float* lno_s    = (const float*)d_in[13];
    const float* lno_b    = (const float*)d_in[14];

    const size_t S = (size_t)BT * DD;       // 2,097,152 elements
    // ---- f32 region ----
    float* ws   = (float*)d_ws;
    float* hbuf = ws;                       // [BT,D]
    float* qkv  = hbuf + S;                 // [3,BT,D]
    float* nbuf = qkv + 3 * S;              // [BT,D]
    float* tmp  = nbuf + S;                 // [BT,D]
    float* comb = tmp + S;                  // [BT,E]
    float* q = qkv, *kk = qkv + S, *vv = qkv + 2 * S;
    // ---- bf16 region (16B aligned: starts on float boundary) ----
    __bf16* bws   = (__bf16*)(comb + (size_t)BT * EE + 8);
    __bf16* xb    = bws;                    // [BT,D]
    __bf16* nbf   = xb + S;                 // [BT,D]
    __bf16* qb    = nbf + S;                // [BT,D]
    __bf16* kb    = qb + S;                 // [BT,D]
    __bf16* vt    = kb + S;                 // [B,NH,hd,T]
    __bf16* w1t_q = vt + S;                 // [3*E, 2H, D]
    __bf16* w2t_q = w1t_q + (size_t)3 * EE * HH2 * DD;   // [3*E, D, 2H]
    __bf16* w1t_b = w2t_q + (size_t)3 * EE * DD * HH2;   // [L*E, 2H, D]
    __bf16* w2t_b = w1t_b + (size_t)6 * EE * HH2 * DD;   // [L*E, D, 2H]
    __bf16* qkvwt = w2t_b + (size_t)6 * EE * DD * HH2;   // [L*3, D, D]
    __bf16* outwt = qkvwt + (size_t)6 * 3 * DD * DD;     // [O, D]

    dim3 b128(128), b256(256);
    const int ATT_GRID = BB * NHD * (TT / 16);   // 2048
    const int ADD_GRID = (int)(S / 256);
    const int EL_GRID  = (int)(S / 256);

    // ---- one-time repacks (per launch; deterministic) ----
    tcvt_kernel<<<dim3((DD * HH2 + 255) / 256, 3 * EE), b256, 0, stream>>>(w1_qkv, w1t_q, DD, HH2);
    tcvt_kernel<<<dim3((HH2 * DD + 255) / 256, 3 * EE), b256, 0, stream>>>(w2_qkv, w2t_q, HH2, DD);
    tcvt_kernel<<<dim3((DD * HH2 + 255) / 256, 6 * EE), b256, 0, stream>>>(w1_b, w1t_b, DD, HH2);
    tcvt_kernel<<<dim3((HH2 * DD + 255) / 256, 6 * EE), b256, 0, stream>>>(w2_b, w2t_b, HH2, DD);
    tcvt_kernel<<<dim3((DD * DD + 255) / 256, 6 * 3), b256, 0, stream>>>(qkv_w, qkvwt, DD, DD);
    tcvt_kernel<<<dim3((DD * OO + 255) / 256, 1), b256, 0, stream>>>(out_w, outwt, DD, OO);
    cvt_kernel<<<EL_GRID, b256, 0, stream>>>(x, xb);

    // ---- top-level MoE for q, k, v ----
    for (int j = 0; j < 3; ++j) {
        moe_combine_kernel<<<BT / 8, b256, 0, stream>>>(
            x, gate_qkv + (size_t)j * DD * EE, comb);
        moe_expert_kernel<<<BT / 16, b128, 0, stream>>>(
            xb, w1t_q + (size_t)j * EE * HH2 * DD,
            w2t_q + (size_t)j * EE * DD * HH2, comb, qkv + (size_t)j * S);
    }
    pack_attn_kernel<<<EL_GRID, b256, 0, stream>>>(q, kk, vv, qb, kb, vt);
    attention_kernel<<<ATT_GRID, b128, 0, stream>>>(qb, kb, vt, tmp);
    add_kernel<<<ADD_GRID, b256, 0, stream>>>(tmp, x, hbuf, (int)S);

    for (int l = 0; l < 6; ++l) {
        ln_kernel<<<BT / 8, b256, 0, stream>>>(hbuf, ln1_s + l * DD, ln1_b + l * DD, nbuf, nbf);
        qkv_gemm_kernel<<<dim3(BT / 16, DD / 64, 3), b128, 0, stream>>>(
            nbf, qkvwt + (size_t)l * 3 * DD * DD, qkv);
        pack_attn_kernel<<<EL_GRID, b256, 0, stream>>>(q, kk, vv, qb, kb, vt);
        attention_kernel<<<ATT_GRID, b128, 0, stream>>>(qb, kb, vt, tmp);
        add_kernel<<<ADD_GRID, b256, 0, stream>>>(hbuf, tmp, hbuf, (int)S);

        ln_kernel<<<BT / 8, b256, 0, stream>>>(hbuf, ln2_s + l * DD, ln2_b + l * DD, nbuf, nbf);
        moe_combine_kernel<<<BT / 8, b256, 0, stream>>>(
            nbuf, gate_b + (size_t)l * DD * EE, comb);
        moe_expert_kernel<<<BT / 16, b128, 0, stream>>>(
            nbf, w1t_b + (size_t)l * EE * HH2 * DD,
            w2t_b + (size_t)l * EE * DD * HH2, comb, tmp);
        add_kernel<<<ADD_GRID, b256, 0, stream>>>(hbuf, tmp, hbuf, (int)S);
    }

    final_head_kernel<<<1, b128, 0, stream>>>(hbuf, outwt, lno_s, lno_b, (float*)d_out);
}
